// MASKTTransformerLayer_10196252361373
// MI455X (gfx1250) — matrix operand order repaired and checked by reading, not compiled, not run on hardware
//
#include <hip/hip_runtime.h>

typedef __attribute__((ext_vector_type(16))) _Float16 v16h;
typedef __attribute__((ext_vector_type(8)))  _Float16 v8h;
typedef __attribute__((ext_vector_type(8)))  float    v8f;

#define NEGV (-1e32f)

// ---------------------------------------------------------------------------
// 128 x (TN*32) GEMM core with double-buffered LDS pipeline.
// C[M,N] = epilogue(A[M,K] x B[K,N]); shapes divide tiles exactly (no guards).
// 256 threads = 8 waves arranged 4(M) x 2(N); wave tile = 32 x (TN*16):
//   TN=4 -> block 128x128, 8 WMMAs/wave/K-step
//   TN=2 -> block 128x64,  4 WMMAs/wave/K-step (used for P@V, N=64)
// LDS: A as [row][k] stride 40 halves; B transposed as [n][k] stride 40, so
// every fragment is a pair of 16B ds_load_b128. One barrier per K-step;
// global loads for step k+1 prefetch into registers during step k's WMMAs.
// ---------------------------------------------------------------------------
template <int TN, bool AF32, bool BF32, bool BIAS, bool GELU, bool RES,
          bool STF, bool STH>
__device__ __forceinline__ void gemm_core(
    const void* __restrict__ A, long lda,
    const void* __restrict__ B, long ldb,
    const float* __restrict__ bias,
    const float* __restrict__ residual, long ldres,
    float* __restrict__ Cf, _Float16* __restrict__ Ch, long ldc,
    int K, int blockM, int blockN)
{
  constexpr int BN   = TN * 32;            // block N: 64 or 128
  constexpr int BVEC = BN / 64;            // v8h per thread for B stage: 1 or 2

  __shared__ _Float16 lA[2][128 * 40];     // [buf][row][k]
  __shared__ _Float16 lBt[2][BN * 40];     // [buf][n][k] (transposed)

  const int t    = threadIdx.x;
  const int lane = t & 31;
  const int wid  = t >> 5;
  const int wm   = (wid >> 1) * 32;
  const int wn   = (wid & 1) * (TN * 16);
  const int r    = lane & 15;
  const int hi   = lane >> 4;

  // staging thread maps
  const int arow = t >> 1, ak = (t & 1) * 16;               // A: 16 elems
  const int bn   = (BN == 64) ? (t & 63) : (t >> 1);        // B column
  const int bks  = (BN == 64) ? ((t >> 6) * 8) : ((t & 1) * 16);

  v8f acc[2][TN];
#pragma unroll
  for (int i = 0; i < 2; ++i)
#pragma unroll
    for (int j = 0; j < TN; ++j)
      acc[i][j] = (v8f){0.f, 0.f, 0.f, 0.f, 0.f, 0.f, 0.f, 0.f};

  v8h aR0, aR1, bR[BVEC];

  auto loadG = [&](int k0) {
    // A tile row: 16 contiguous elements
    const long abase = (long)(blockM + arow) * lda + (k0 + ak);
    if (AF32) {
      const float* p = (const float*)A + abase;
#pragma unroll
      for (int j = 0; j < 8; ++j) {
        aR0[j] = (_Float16)p[j];
        aR1[j] = (_Float16)p[8 + j];
      }
    } else {
      const v8h* p = (const v8h*)((const _Float16*)A + abase);
      aR0 = p[0];
      aR1 = p[1];
    }
    // B tile column slice: BVEC*8 k-values at fixed n
    const long gn = blockN + bn;
#pragma unroll
    for (int v = 0; v < BVEC; ++v)
#pragma unroll
      for (int kk = 0; kk < 8; ++kk) {
        const long idx = (long)(k0 + bks + v * 8 + kk) * ldb + gn;
        bR[v][kk] = BF32 ? (_Float16)((const float*)B)[idx]
                         : ((const _Float16*)B)[idx];
      }
  };
  auto storeL = [&](int buf) {
    *(v8h*)&lA[buf][arow * 40 + ak]     = aR0;
    *(v8h*)&lA[buf][arow * 40 + ak + 8] = aR1;
#pragma unroll
    for (int v = 0; v < BVEC; ++v)
      *(v8h*)&lBt[buf][bn * 40 + bks + v * 8] = bR[v];
  };

  const int nk = K >> 5;
  loadG(0);
  storeL(0);

  for (int ki = 0; ki < nk; ++ki) {
    __syncthreads();
    const int cur = ki & 1;

    // prefetch next K-step into registers (overlaps with WMMAs below)
    if (ki + 1 < nk) loadG((ki + 1) << 5);

    // fragments: each is two aligned v8h (ds_load_b128)
    v16h af[2], bf[TN];
#pragma unroll
    for (int tm = 0; tm < 2; ++tm) {
      const int row = wm + tm * 16 + r;
      const v8h lo = *(const v8h*)&lA[cur][row * 40 + hi * 8];
      const v8h hh = *(const v8h*)&lA[cur][row * 40 + 16 + hi * 8];
#pragma unroll
      for (int e = 0; e < 8; ++e) { af[tm][e] = lo[e]; af[tm][8 + e] = hh[e]; }
    }
#pragma unroll
    for (int tn = 0; tn < TN; ++tn) {
      const int col = wn + tn * 16 + r;
      const v8h lo = *(const v8h*)&lBt[cur][col * 40 + hi * 8];
      const v8h hh = *(const v8h*)&lBt[cur][col * 40 + 16 + hi * 8];
#pragma unroll
      for (int e = 0; e < 8; ++e) { bf[tn][e] = lo[e]; bf[tn][8 + e] = hh[e]; }
    }

#pragma unroll
    for (int tm = 0; tm < 2; ++tm)
#pragma unroll
      for (int tn = 0; tn < TN; ++tn)
        acc[tm][tn] = __builtin_amdgcn_wmma_f32_16x16x32_f16(
            false, af[tm], false, bf[tn], (short)0, acc[tm][tn], false, false);

    // stash prefetched data into the other buffer (safe: its readers finished
    // before the barrier at the top of this iteration)
    if (ki + 1 < nk) storeL((ki + 1) & 1);
  }

  // ---- epilogue: bias -> gelu -> residual -> store ----
  const int rbase = hi * 8;
#pragma unroll
  for (int tm = 0; tm < 2; ++tm)
#pragma unroll
    for (int tn = 0; tn < TN; ++tn) {
      const int gn = blockN + wn + tn * 16 + r;
      const float bv = BIAS ? bias[gn] : 0.f;
#pragma unroll
      for (int e = 0; e < 8; ++e) {
        const int gm = blockM + wm + tm * 16 + rbase + e;
        float v = acc[tm][tn][e];
        if (BIAS) v += bv;
        if (GELU) v = 0.5f * v * (1.f + erff(v * 0.70710678118f));
        if (RES)  v += residual[(long)gm * ldres + gn];
        if (STF)  Cf[(long)gm * ldc + gn] = v;
        if (STH)  Ch[(long)gm * ldc + gn] = (_Float16)v;
      }
    }
}

template <int TN, bool AF32, bool BF32, bool BIAS, bool GELU, bool RES,
          bool STF, bool STH>
__global__ __launch_bounds__(256)
void gemm_kernel(const void* __restrict__ A, long lda,
                 const void* __restrict__ B, long ldb,
                 const float* __restrict__ bias,
                 const float* __restrict__ residual, long ldres,
                 float* __restrict__ Cf, _Float16* __restrict__ Ch, long ldc,
                 int K)
{
  gemm_core<TN, AF32, BF32, BIAS, GELU, RES, STF, STH>(
      A, lda, B, ldb, bias, residual, ldres, Cf, Ch, ldc, K,
      blockIdx.x * 128, blockIdx.y * (TN * 32));
}

// P[1024,1024] (f32, in d_out) x V[1024,64] (f16 head slice) -> concat f16
__global__ __launch_bounds__(256)
void gemm_pv_kernel(const float* __restrict__ P, const _Float16* __restrict__ Vp,
                    _Float16* __restrict__ concat)
{
  const int z = blockIdx.z, b = z >> 3, h = z & 7;
  const float* A = P + (long)z * 1024 * 1024;
  const _Float16* B = Vp + (long)b * 1024 * 512 + h * 64;
  _Float16* C = concat + (long)b * 1024 * 512 + h * 64;
  gemm_core<2, true, false, false, false, false, false, true>(
      A, 1024, B, 512, nullptr, nullptr, 0, nullptr, C, 512,
      1024, blockIdx.x * 128, 0);
}

// ---------------------------------------------------------------------------
// Raw attention scores: S = Q Kt / sqrt(64) per (b,h). 128x128 block tile,
// 8 waves arranged 4(M) x 2(N), each 32x64 = 2x4 WMMA accumulators.
// K=64 (2 steps) so no pipeline needed; all fragments are 16B ds loads.
// ---------------------------------------------------------------------------
__global__ __launch_bounds__(256)
void attn_scores_kernel(const _Float16* __restrict__ Qp,
                        const _Float16* __restrict__ Kp,
                        float* __restrict__ attn)
{
  __shared__ _Float16 lQ[128 * 40];
  __shared__ _Float16 lK[128 * 40];

  const int z = blockIdx.z, b = z >> 3, h = z & 7;
  const _Float16* Qb = Qp + (long)b * 1024 * 512 + h * 64;
  const _Float16* Kb = Kp + (long)b * 1024 * 512 + h * 64;
  const int blockM = blockIdx.x * 128;
  const int blockN = blockIdx.y * 128;

  const int t    = threadIdx.x;
  const int lane = t & 31;
  const int wid  = t >> 5;
  const int wm   = (wid >> 1) * 32;
  const int wn   = (wid & 1) * 64;
  const int r    = lane & 15;
  const int hi   = lane >> 4;

  v8f acc[2][4];
#pragma unroll
  for (int i = 0; i < 2; ++i)
#pragma unroll
    for (int j = 0; j < 4; ++j)
      acc[i][j] = (v8f){0.f, 0.f, 0.f, 0.f, 0.f, 0.f, 0.f, 0.f};

  const int arow = t >> 1, ak = (t & 1) * 16;

  for (int k0 = 0; k0 < 64; k0 += 32) {
    {
      const v8h* q = (const v8h*)&Qb[(long)(blockM + arow) * 512 + k0 + ak];
      const v8h* k = (const v8h*)&Kb[(long)(blockN + arow) * 512 + k0 + ak];
      *(v8h*)&lQ[arow * 40 + ak]     = q[0];
      *(v8h*)&lQ[arow * 40 + ak + 8] = q[1];
      *(v8h*)&lK[arow * 40 + ak]     = k[0];
      *(v8h*)&lK[arow * 40 + ak + 8] = k[1];
    }
    __syncthreads();

    v16h af[2], bf[4];
#pragma unroll
    for (int tm = 0; tm < 2; ++tm) {
      const int row = wm + tm * 16 + r;
      const v8h lo = *(const v8h*)&lQ[row * 40 + hi * 8];
      const v8h hh = *(const v8h*)&lQ[row * 40 + 16 + hi * 8];
#pragma unroll
      for (int e = 0; e < 8; ++e) { af[tm][e] = lo[e]; af[tm][8 + e] = hh[e]; }
    }
#pragma unroll
    for (int tn = 0; tn < 4; ++tn) {
      const int col = wn + tn * 16 + r;   // B[k][n] = K[n][k]
      const v8h lo = *(const v8h*)&lK[col * 40 + hi * 8];
      const v8h hh = *(const v8h*)&lK[col * 40 + 16 + hi * 8];
#pragma unroll
      for (int e = 0; e < 8; ++e) { bf[tn][e] = lo[e]; bf[tn][8 + e] = hh[e]; }
    }
#pragma unroll
    for (int tm = 0; tm < 2; ++tm)
#pragma unroll
      for (int tn = 0; tn < 4; ++tn)
        acc[tm][tn] = __builtin_amdgcn_wmma_f32_16x16x32_f16(
            false, af[tm], false, bf[tn], (short)0, acc[tm][tn], false, false);
    __syncthreads();
  }

  float* out = attn + (long)z * 1024 * 1024;
  const int rbase = hi * 8;
#pragma unroll
  for (int tm = 0; tm < 2; ++tm)
#pragma unroll
    for (int tn = 0; tn < 4; ++tn) {
      const int gn = blockN + wn + tn * 16 + r;
#pragma unroll
      for (int e = 0; e < 8; ++e) {
        const int gm = blockM + wm + tm * 16 + rbase + e;
        out[(long)gm * 1024 + gn] = acc[tm][tn][e] * 0.125f;  // 1/sqrt(64)
      }
    }
}

// ---------------------------------------------------------------------------
// Per-row: masked softmax -> cumsum -> dist effect -> second softmax (in place)
// One 1024-thread block per row (one element per thread).
// ---------------------------------------------------------------------------
__global__ __launch_bounds__(1024)
void attn_row_kernel(float* __restrict__ attn, const float* __restrict__ gammas,
                     const int* __restrict__ maskp)
{
  __shared__ float buf[1024];
  const int row = blockIdx.x;            // bh*1024 + i
  const int i = row & 1023;
  const int h = (row >> 10) & 7;
  const int j = threadIdx.x;
  float* rp = attn + (long)row * 1024;

  const float s = rp[j];
  const int mv = *maskp;
  const bool mm = (mv == 0 || mv == 1);
  const bool allowed = (!mm) || (j < i + mv);

  // ---- softmax 1 ----
  float x1 = allowed ? s : NEGV;
  buf[j] = x1; __syncthreads();
  for (int off = 512; off > 0; off >>= 1) {
    if (j < off) buf[j] = fmaxf(buf[j], buf[j + off]);
    __syncthreads();
  }
  const float mx1 = buf[0]; __syncthreads();
  const float e1 = expf(x1 - mx1);
  buf[j] = e1; __syncthreads();
  for (int off = 512; off > 0; off >>= 1) {
    if (j < off) buf[j] += buf[j + off];
    __syncthreads();
  }
  const float sum1 = buf[0]; __syncthreads();
  const float s1 = allowed ? (e1 / sum1) : 0.f;

  // ---- inclusive scan (cumsum) ----
  buf[j] = s1; __syncthreads();
  for (int off = 1; off < 1024; off <<= 1) {
    const float add = (j >= off) ? buf[j - off] : 0.f;
    __syncthreads();
    buf[j] += add;
    __syncthreads();
  }
  const float c = buf[j];
  const float T = buf[1023];
  __syncthreads();

  // ---- distance effect ----
  const float pe = fabsf((float)i - (float)j);
  const float dist = sqrtf(fmaxf((T - c) * pe, 0.f));
  const float g = gammas[h];
  const float sp = (g > 20.f) ? g : log1pf(expf(g));  // softplus
  float eff = expf(-sp * dist);
  eff = fminf(fmaxf(eff, 1e-5f), 1e5f);

  // ---- softmax 2 ----
  const float x2 = allowed ? (s * eff) : NEGV;
  buf[j] = x2; __syncthreads();
  for (int off = 512; off > 0; off >>= 1) {
    if (j < off) buf[j] = fmaxf(buf[j], buf[j + off]);
    __syncthreads();
  }
  const float mx2 = buf[0]; __syncthreads();
  const float e2 = expf(x2 - mx2);
  buf[j] = e2; __syncthreads();
  for (int off = 512; off > 0; off >>= 1) {
    if (j < off) buf[j] += buf[j + off];
    __syncthreads();
  }
  rp[j] = e2 / buf[0];
}

// ---------------------------------------------------------------------------
// LayerNorm over rows of 512; 256 threads, 2 elements each.
// ---------------------------------------------------------------------------
__global__ __launch_bounds__(256)
void layernorm_kernel(const float* __restrict__ in,
                      const float* __restrict__ scale,
                      const float* __restrict__ bias,
                      float* __restrict__ out)
{
  __shared__ float red[256];
  const int row = blockIdx.x;
  const int t = threadIdx.x;
  const float* rp = in + (long)row * 512;
  const float a = rp[t], b = rp[t + 256];

  red[t] = a + b; __syncthreads();
  for (int off = 128; off > 0; off >>= 1) {
    if (t < off) red[t] += red[t + off];
    __syncthreads();
  }
  const float mu = red[0] * (1.f / 512.f); __syncthreads();
  const float da = a - mu, db = b - mu;
  red[t] = da * da + db * db; __syncthreads();
  for (int off = 128; off > 0; off >>= 1) {
    if (t < off) red[t] += red[t + off];
    __syncthreads();
  }
  const float inv = rsqrtf(red[0] * (1.f / 512.f) + 1e-5f);
  float* op = out + (long)row * 512;
  op[t]       = da * inv * scale[t]       + bias[t];
  op[t + 256] = db * inv * scale[t + 256] + bias[t + 256];
}

// ---------------------------------------------------------------------------
extern "C" void kernel_launch(void* const* d_in, const int* in_sizes, int n_in,
                              void* d_out, int out_size, void* d_ws, size_t ws_size,
                              hipStream_t stream)
{
  const int*   maskp  = (const int*)d_in[0];
  const float* query  = (const float*)d_in[1];
  const float* key    = (const float*)d_in[2];
  const float* values = (const float*)d_in[3];
  const float* Wq = (const float*)d_in[4];  const float* bq = (const float*)d_in[5];
  const float* Wk = (const float*)d_in[6];  const float* bk = (const float*)d_in[7];
  const float* Wv = (const float*)d_in[8];  const float* bv = (const float*)d_in[9];
  const float* Wo = (const float*)d_in[10]; const float* bo = (const float*)d_in[11];
  const float* gammas = (const float*)d_in[12];
  const float* ln1s = (const float*)d_in[13]; const float* ln1b = (const float*)d_in[14];
  const float* W1 = (const float*)d_in[15]; const float* b1 = (const float*)d_in[16];
  const float* W2 = (const float*)d_in[17]; const float* b2 = (const float*)d_in[18];
  const float* ln2s = (const float*)d_in[19]; const float* ln2b = (const float*)d_in[20];

  const long MTOK = 8192;  // BS*SEQ
  float* xout = (float*)d_out;                       // [8,1024,512]
  float* attn = xout + (size_t)MTOK * 512;           // [8,8,1024,1024]

  char* ws = (char*)d_ws;
  size_t off = 0;
  auto walloc = [&](size_t bytes) -> void* {
    void* p = ws + off;
    off += (bytes + 255) & ~(size_t)255;
    return p;
  };
  _Float16* Qp     = (_Float16*)walloc(MTOK * 512 * sizeof(_Float16));
  _Float16* Kp     = (_Float16*)walloc(MTOK * 512 * sizeof(_Float16));
  _Float16* Vp     = (_Float16*)walloc(MTOK * 512 * sizeof(_Float16));
  _Float16* concat = (_Float16*)walloc(MTOK * 512 * sizeof(_Float16));
  float*    resid1 = (float*)walloc(MTOK * 512 * sizeof(float));
  float*    x      = (float*)walloc(MTOK * 512 * sizeof(float));
  _Float16* ffh    = (_Float16*)walloc(MTOK * 2048 * sizeof(_Float16));
  float*    resid2 = (float*)walloc(MTOK * 512 * sizeof(float));

  const dim3 blk(256);

  // Q/K/V projections: f32 in, f32 weights, +bias, store f16 [8192,512]
  gemm_kernel<4, true, true, true, false, false, false, true>
      <<<dim3(64, 4, 1), blk, 0, stream>>>(
          query, 512, Wq, 512, bq, nullptr, 0, nullptr, Qp, 512, 512);
  gemm_kernel<4, true, true, true, false, false, false, true>
      <<<dim3(64, 4, 1), blk, 0, stream>>>(
          key, 512, Wk, 512, bk, nullptr, 0, nullptr, Kp, 512, 512);
  gemm_kernel<4, true, true, true, false, false, false, true>
      <<<dim3(64, 4, 1), blk, 0, stream>>>(
          values, 512, Wv, 512, bv, nullptr, 0, nullptr, Vp, 512, 512);

  // raw scores (scaled) straight into d_out attn region
  attn_scores_kernel<<<dim3(8, 8, 64), blk, 0, stream>>>(Qp, Kp, attn);

  // per-row masked softmax / cumsum-dist / second softmax, in place
  attn_row_kernel<<<dim3(64 * 1024), dim3(1024), 0, stream>>>(attn, gammas, maskp);

  // out = P @ V -> concat f16 [8192,512]   (N=64 -> TN=2 tile)
  gemm_pv_kernel<<<dim3(8, 1, 64), blk, 0, stream>>>(attn, Vp, concat);

  // query2 = concat @ Wo + bo ; resid1 = query + query2  (A f16, B f32)
  gemm_kernel<4, false, true, true, false, true, true, false>
      <<<dim3(64, 4, 1), blk, 0, stream>>>(
          concat, 512, Wo, 512, bo, query, 512, resid1, nullptr, 512, 512);

  // x = LN1(resid1)
  layernorm_kernel<<<dim3(8192), blk, 0, stream>>>(resid1, ln1s, ln1b, x);

  // ffh = gelu(x @ W1 + b1) -> f16 [8192,2048]  (A f32, B f32, GELU)
  gemm_kernel<4, true, true, true, true, false, false, true>
      <<<dim3(64, 16, 1), blk, 0, stream>>>(
          x, 512, W1, 2048, b1, nullptr, 0, nullptr, ffh, 2048, 512);

  // resid2 = x + (ffh @ W2 + b2)  (A f16, B f32, residual)
  gemm_kernel<4, false, true, true, false, true, true, false>
      <<<dim3(64, 4, 1), blk, 0, stream>>>(
          ffh, 2048, W2, 512, b2, x, 512, resid2, nullptr, 512, 2048);

  // x_out = LN2(resid2) -> d_out
  layernorm_kernel<<<dim3(8192), blk, 0, stream>>>(resid2, ln2s, ln2b, xout);
}
